// TextureTransfer_44676249813712
// MI455X (gfx1250) — matrix operand order repaired and checked by reading, not compiled, not run on hardware
//
#include <hip/hip_runtime.h>

// ---------- CDNA5 WMMA types ----------
typedef __attribute__((ext_vector_type(16))) __bf16 v16bf;
typedef __attribute__((ext_vector_type(8)))  float  v8f;

#define BATCH   2
#define HW      64
#define NPIX    4096      // 64*64
#define CCH     256
#define KDIM    2304      // 256*9
#define NCHUNK  72        // KDIM/32
#define NTILE   256       // NPIX/16
#define FRAG_ELEMS 512    // 32 lanes * 16 bf16

// workspace byte offsets
#define WS_SSQ_LR   0
#define WS_SSQ_RS   32768
#define WS_INV_LR   65536
#define WS_INV_RS   98304
#define WS_RARG     131072
#define WS_APACK    163840UL
#define APACK_BYTES 37748736UL   // 2*4096*2304*2
#define WS_BPACK    (WS_APACK + APACK_BYTES)

__device__ __forceinline__ unsigned short f2bf(float f) {
  unsigned int u = __float_as_uint(f);
  unsigned int r = u + 0x7FFFu + ((u >> 16) & 1u);   // round-to-nearest-even
  return (unsigned short)(r >> 16);
}

// ---------- 1) per-pixel channel sum-of-squares for both feature maps ----------
__global__ void ssq_kernel(const float* __restrict__ lr, const float* __restrict__ rs,
                           float* __restrict__ ssq_lr, float* __restrict__ ssq_rs) {
  int t = blockIdx.x * blockDim.x + threadIdx.x;
  if (t >= 2 * BATCH * NPIX) return;
  int which = t / (BATCH * NPIX);
  int rem   = t % (BATCH * NPIX);
  int b = rem / NPIX, p = rem % NPIX;
  const float* src = which ? rs : lr;
  const float* base = src + (size_t)b * CCH * NPIX + p;
  float s = 0.f;
  for (int c = 0; c < CCH; ++c) { float v = base[(size_t)c * NPIX]; s += v * v; }
  (which ? ssq_rs : ssq_lr)[b * NPIX + p] = s;
}

// ---------- 2) 3x3 neighborhood sum -> 1/max(norm, eps) ----------
__global__ void invnorm_kernel(const float* __restrict__ ssq_lr, const float* __restrict__ ssq_rs,
                               float* __restrict__ inv_lr, float* __restrict__ inv_rs) {
  int t = blockIdx.x * blockDim.x + threadIdx.x;
  if (t >= 2 * BATCH * NPIX) return;
  int which = t / (BATCH * NPIX);
  int rem   = t % (BATCH * NPIX);
  int b = rem / NPIX, p = rem % NPIX;
  int y = p >> 6, x = p & 63;
  const float* s = (which ? ssq_rs : ssq_lr) + b * NPIX;
  float acc = 0.f;
  for (int dy = -1; dy <= 1; ++dy) {
    int yy = y + dy; if (yy < 0 || yy >= HW) continue;
    for (int dx = -1; dx <= 1; ++dx) {
      int xx = x + dx; if (xx < 0 || xx >= HW) continue;
      acc += s[yy * HW + xx];
    }
  }
  float n = fmaxf(sqrtf(acc), 1e-12f);
  (which ? inv_rs : inv_lr)[b * NPIX + p] = 1.0f / n;
}

// ---------- 3a) pack normalized refsr patches into WMMA A-fragment layout ----------
// A layout (16-bit 16x32): lane(0-15)=row M, vec elem e -> K = 16*half + e
__global__ void packA_kernel(const float* __restrict__ rs, const float* __restrict__ inv_rs,
                             unsigned short* __restrict__ Ap) {
  int t = blockIdx.x * 256 + threadIdx.x;           // BATCH*NTILE*NCHUNK*32 threads
  int lane = t & 31;
  int rest = t >> 5;
  int ch = rest % NCHUNK; rest /= NCHUNK;
  int rt = rest % NTILE;  int b  = rest / NTILE;
  int m = lane & 15, hf = lane >> 4;
  int r = rt * 16 + m;
  int py = r >> 6, px = r & 63;
  float invn = inv_rs[b * NPIX + r];
  const float* src = rs + (size_t)b * CCH * NPIX;
  unsigned short vals[16];
#pragma unroll
  for (int e = 0; e < 16; ++e) {
    int kl = 16 * hf + e;
    int k = ch * 32 + kl;
    int c = k / 9, idx = k % 9;
    int i = idx / 3, j = idx % 3;
    int y = py + i - 1, x = px + j - 1;
    float v = 0.f;
    if (y >= 0 && y < HW && x >= 0 && x < HW) v = src[(size_t)c * NPIX + y * HW + x] * invn;
    vals[e] = f2bf(v);
  }
  size_t off = ((size_t)((b * NTILE + rt) * NCHUNK + ch)) * FRAG_ELEMS + lane * 16;
  uint4* dst = (uint4*)(Ap + off);
  dst[0] = *(uint4*)&vals[0];
  dst[1] = *(uint4*)&vals[8];
}

// ---------- 3b) pack normalized lrsr patches into WMMA B-fragment layout ----------
// B layout (16-bit 32x16): lane(0-15)=col N, vec elem e -> K = 16*half + e
__global__ void packB_kernel(const float* __restrict__ lr, const float* __restrict__ inv_lr,
                             unsigned short* __restrict__ Bp) {
  int t = blockIdx.x * 256 + threadIdx.x;
  int lane = t & 31;
  int rest = t >> 5;
  int ch = rest % NCHUNK; rest /= NCHUNK;
  int ct = rest % NTILE;  int b  = rest / NTILE;
  int n = lane & 15, hf = lane >> 4;
  int col = ct * 16 + n;
  int py = col >> 6, px = col & 63;
  float invn = inv_lr[b * NPIX + col];
  const float* src = lr + (size_t)b * CCH * NPIX;
  unsigned short vals[16];
#pragma unroll
  for (int e = 0; e < 16; ++e) {
    int k = ch * 32 + 16 * hf + e;
    int c = k / 9, idx = k % 9;
    int i = idx / 3, j = idx % 3;
    int y = py + i - 1, x = px + j - 1;
    float v = 0.f;
    if (y >= 0 && y < HW && x >= 0 && x < HW) v = src[(size_t)c * NPIX + y * HW + x] * invn;
    vals[e] = f2bf(v);
  }
  size_t off = ((size_t)((b * NTILE + ct) * NCHUNK + ch)) * FRAG_ELEMS + lane * 16;
  uint4* dst = (uint4*)(Bp + off);
  dst[0] = *(uint4*)&vals[0];
  dst[1] = *(uint4*)&vals[8];
}

// ---------- 4) fused GEMM + column-argmax ----------
// Block = 8 waves. Block owns 32 columns (2 col-tiles); B panel (144KB) staged in LDS.
// Each wave sweeps 32 row-tiles, 72 K-chunks each, 2 WMMAs per A fragment.
__global__ void __launch_bounds__(256) gemm_argmax_kernel(const unsigned short* __restrict__ Ap,
                                                          const unsigned short* __restrict__ Bp,
                                                          float* __restrict__ S_out,
                                                          int* __restrict__ Rarg) {
  extern __shared__ char smem[];
  unsigned short* bsm = (unsigned short*)smem;                 // 2*72*512 ushorts = 147456 B
  float* redv = (float*)(smem + 147456);                       // 32 cols * 16 slots
  int*   redi = (int*)  (smem + 147456 + 32 * 16 * 4);

  int blk = blockIdx.x;           // BATCH * 128
  int b  = blk >> 7;
  int cg = blk & 127;
  int ct0 = cg * 2;

  // cooperative copy of the 32-column B panel into LDS (contiguous in packed layout)
  {
    const uint4* src = (const uint4*)(Bp + ((size_t)(b * NTILE + ct0) * NCHUNK) * FRAG_ELEMS);
    uint4* dst = (uint4*)bsm;
    const int n128 = (2 * NCHUNK * FRAG_ELEMS) / 8;            // 9216 uint4
    for (int i = threadIdx.x; i < n128; i += 256) dst[i] = src[i];
  }
  __syncthreads();

  int lane = threadIdx.x & 31;
  int w    = threadIdx.x >> 5;
  int hf   = lane >> 4;

  float bv0 = -3.4e38f, bv1 = -3.4e38f;
  int   bi0 = 0,        bi1 = 0;

  const __bf16* A  = (const __bf16*)Ap;
  const __bf16* BL = (const __bf16*)bsm;

  for (int rt = w; rt < NTILE; rt += 8) {
    v8f c0 = {}; v8f c1 = {};
    const __bf16* arow = A + ((size_t)((b * NTILE + rt) * NCHUNK)) * FRAG_ELEMS + lane * 16;
    const __bf16* b0p  = BL + lane * 16;
    const __bf16* b1p  = BL + (size_t)NCHUNK * FRAG_ELEMS + lane * 16;
    for (int ch = 0; ch < NCHUNK; ++ch) {
      v16bf a   = *(const v16bf*)(arow); arow += FRAG_ELEMS;
      v16bf bb0 = *(const v16bf*)(b0p);  b0p  += FRAG_ELEMS;
      v16bf bb1 = *(const v16bf*)(b1p);  b1p  += FRAG_ELEMS;
      c0 = __builtin_amdgcn_wmma_f32_16x16x32_bf16(false, a, false, bb0, (short)0, c0, false, false);
      c1 = __builtin_amdgcn_wmma_f32_16x16x32_bf16(false, a, false, bb1, (short)0, c1, false, false);
    }
    // C layout: VGPR g -> M = g + 8*hf ; running argmax over rows r
    int rbase = rt * 16 + hf * 8;
#pragma unroll
    for (int g = 0; g < 8; ++g) {
      float v0 = c0[g]; if (v0 > bv0) { bv0 = v0; bi0 = rbase + g; }
      float v1 = c1[g]; if (v1 > bv1) { bv1 = v1; bi1 = rbase + g; }
    }
  }

  // lane-half + wave reduction via LDS: 16 candidates per column
  int n = lane & 15;
  int slot = w * 2 + hf;
  redv[n * 16 + slot]        = bv0;  redi[n * 16 + slot]        = bi0;
  redv[(16 + n) * 16 + slot] = bv1;  redi[(16 + n) * 16 + slot] = bi1;
  __syncthreads();

  if (threadIdx.x < 32) {
    int col = threadIdx.x;
    float bv = redv[col * 16]; int bi = redi[col * 16];
    for (int s2 = 1; s2 < 16; ++s2) {
      float v = redv[col * 16 + s2];
      if (v > bv) { bv = v; bi = redi[col * 16 + s2]; }
    }
    int gcol = cg * 32 + col;
    S_out[b * NPIX + gcol] = bv;     // R_star -> S
    Rarg [b * NPIX + gcol] = bi;
  }
}

// ---------- 5) gather + fold (each output pixel = 1/9 * sum of its 9 covering patches) ----------
// For output (y,x): ly = floor((y+pad)/s) - ty (ty=0..2), i = y+pad - s*ly in [0,k).
// Patch p = Rarg[ly*64+lx]; contribution = ref[c, s*py + i - pad, s*px + j - pad] (0 if OOB).
__global__ void gather_fold_kernel(const float* __restrict__ ref, const int* __restrict__ Rarg,
                                   float* __restrict__ out, int C, int Hout, int pad, int s) {
  size_t t = (size_t)blockIdx.x * 256 + threadIdx.x;
  int x = (int)(t % Hout); size_t r1 = t / Hout;
  int y = (int)(r1 % Hout); size_t r2 = r1 / Hout;
  int c = (int)(r2 % C);    int b = (int)(r2 / C);
  const float* rp = ref + ((size_t)(b * C + c)) * Hout * Hout;
  const int* ra = Rarg + b * NPIX;
  float acc = 0.f;
  int qy = (y + pad) / s;
  int qx = (x + pad) / s;
#pragma unroll
  for (int ty = 0; ty < 3; ++ty) {
    int ly = qy - ty; if (ly < 0 || ly >= HW) continue;
    int i = y + pad - s * ly;
#pragma unroll
    for (int tx = 0; tx < 3; ++tx) {
      int lx = qx - tx; if (lx < 0 || lx >= HW) continue;
      int j = x + pad - s * lx;
      int p = ra[ly * HW + lx];
      int py = p >> 6, px = p & 63;
      int ry = s * py + i - pad, rx = s * px + j - pad;
      if (ry >= 0 && ry < Hout && rx >= 0 && rx < Hout)
        acc += rp[ry * Hout + rx];
    }
  }
  out[t] = acc * (1.0f / 9.0f);
}

extern "C" void kernel_launch(void* const* d_in, const int* in_sizes, int n_in,
                              void* d_out, int out_size, void* d_ws, size_t ws_size,
                              hipStream_t stream) {
  const float* lrsr  = (const float*)d_in[0];   // (2,256,64,64)
  const float* refsr = (const float*)d_in[1];   // (2,256,64,64)
  const float* ref1  = (const float*)d_in[2];   // (2,64,256,256)
  const float* ref2  = (const float*)d_in[3];   // (2,128,128,128)
  const float* ref3  = (const float*)d_in[4];   // (2,256,64,64)
  float* out = (float*)d_out;
  char* ws = (char*)d_ws;

  float* ssq_lr = (float*)(ws + WS_SSQ_LR);
  float* ssq_rs = (float*)(ws + WS_SSQ_RS);
  float* inv_lr = (float*)(ws + WS_INV_LR);
  float* inv_rs = (float*)(ws + WS_INV_RS);
  int*   rarg   = (int*)  (ws + WS_RARG);
  unsigned short* Ap = (unsigned short*)(ws + WS_APACK);
  unsigned short* Bp = (unsigned short*)(ws + WS_BPACK);

  ssq_kernel    <<<(2 * BATCH * NPIX + 255) / 256, 256, 0, stream>>>(lrsr, refsr, ssq_lr, ssq_rs);
  invnorm_kernel<<<(2 * BATCH * NPIX + 255) / 256, 256, 0, stream>>>(ssq_lr, ssq_rs, inv_lr, inv_rs);

  const int packThreads = BATCH * NTILE * NCHUNK * 32;   // 1,179,648
  packA_kernel<<<packThreads / 256, 256, 0, stream>>>(refsr, inv_rs, Ap);
  packB_kernel<<<packThreads / 256, 256, 0, stream>>>(lrsr,  inv_lr, Bp);

  size_t shbytes = 147456 + 32 * 16 * 8;                 // B panel + reduction scratch
  gemm_argmax_kernel<<<BATCH * 128, 256, shbytes, stream>>>(Ap, Bp, out, rarg);

  float* T3 = out + 8192;
  float* T2 = out + 8192 + 2097152;
  float* T1 = out + 8192 + 2097152 + 4194304;
  gather_fold_kernel<<<2097152 / 256, 256, 0, stream>>>(ref3, rarg, T3, 256,  64, 1, 1);
  gather_fold_kernel<<<4194304 / 256, 256, 0, stream>>>(ref2, rarg, T2, 128, 128, 2, 2);
  gather_fold_kernel<<<8388608 / 256, 256, 0, stream>>>(ref1, rarg, T1,  64, 256, 4, 4);
}
